// EquivariantAttentionUpsample_51926154609219
// MI455X (gfx1250) — compile-verified, matrix operand order checked by reading, not run on hardware
//
#include <hip/hip_runtime.h>
#include <hip/hip_fp16.h>
#include <math.h>

typedef _Float16 v16h __attribute__((ext_vector_type(16)));
typedef _Float16 v8h  __attribute__((ext_vector_type(8)));
typedef float    v8f  __attribute__((ext_vector_type(8)));

#define WSRC 128
#define HRUP 512
#define WRUP 512
#define NTOT (HRUP * WRUP)

// ---------------- workspace layout (float offsets) ----------------
#define OFF_W440 0        // (9,9,1)    81
#define OFF_W660 81       // (13,13,1)  169
#define OFF_W404 250      // (9,1,9)    81
#define OFF_W424 331      // (9,5,9)    405
#define OFF_W426 736      // (9,5,13)   585
#define OFF_W606 1321     // (13,1,13)  169
#define OFF_W624 1490     // (13,5,9)   585
#define OFF_W626 2075     // (13,5,13)  845
#define OFF_W444 2920     // (9,9,9)    729
#define OFF_W446 3649     // (9,9,13)   1053
#define OFF_W464 4702     // (9,13,9)   1053
#define OFF_W466 5755     // (9,13,13)  1521
#define OFF_W644 7276     // (13,9,9)   1053
#define OFF_W646 8329     // (13,9,13)  1521
#define OFF_W664 9850     // (13,13,9)  1521
#define OFF_W666 11371    // (13,13,13) 2197
#define OFF_S    13568    // 22x22
#define OFF_MV   14052    // [9][32][32] padded
#define OFF_G    23268    // [512][32]  padded
// total 39652 floats (~155 KB) in d_ws

#define C_SC 0.70710678118654752
#define C4V  1.73205080756887729
#define C6V  2.08166599946613295
#define C4O  1.5
#define C6O  1.80277563773199464

// ================= setup kernel 1: Wigner 3j tensors =================
__device__ double dfact(int n) { double r = 1.0; for (int i = 2; i <= n; ++i) r *= (double)i; return r; }

__device__ double cgcoef(int j1, int j2, int j3, int m1, int m2, int m3) {
  if (m1 + m2 != m3) return 0.0;
  double pre = sqrt((2.0 * j3 + 1.0) * dfact(j1 + j2 - j3) * dfact(j1 - j2 + j3) *
                    dfact(-j1 + j2 + j3) / dfact(j1 + j2 + j3 + 1));
  pre *= sqrt(dfact(j1 + m1) * dfact(j1 - m1) * dfact(j2 + m2) * dfact(j2 - m2) *
              dfact(j3 + m3) * dfact(j3 - m3));
  int kmin = 0;
  if (j2 - j3 - m1 > kmin) kmin = j2 - j3 - m1;
  if (j1 - j3 + m2 > kmin) kmin = j1 - j3 + m2;
  int kmax = j1 + j2 - j3;
  if (j1 - m1 < kmax) kmax = j1 - m1;
  if (j2 + m2 < kmax) kmax = j2 + m2;
  double s = 0.0;
  for (int k = kmin; k <= kmax; ++k) {
    double d = dfact(k) * dfact(j1 + j2 - j3 - k) * dfact(j1 - m1 - k) *
               dfact(j2 + m2 - k) * dfact(j3 - j2 + m1 + k) * dfact(j3 - j1 - m2 + k);
    s += ((k & 1) ? -1.0 : 1.0) / d;
  }
  return pre * s;
}

// U(l) row a: at most 2 nonzero complex entries
__device__ int urow(int l, int a, int col[2], double re[2], double im[2]) {
  const double inv2 = 0.70710678118654752440;
  if (a == l) { col[0] = l; re[0] = 1.0; im[0] = 0.0; return 1; }
  if (a > l) {
    int m = a - l; double sgn = (m & 1) ? -1.0 : 1.0;
    col[0] = l + m; re[0] = sgn * inv2; im[0] = 0.0;
    col[1] = l - m; re[1] = inv2;       im[1] = 0.0;
    return 2;
  }
  int m = l - a; double sgn = (m & 1) ? -1.0 : 1.0;
  col[0] = l - m; re[0] = 0.0; im[0] = inv2;
  col[1] = l + m; re[1] = 0.0; im[1] = -sgn * inv2;
  return 2;
}

__global__ void __launch_bounds__(256) w3j_kernel(float* __restrict__ ws) {
  const int tbl[16][4] = {
    {4,4,0,OFF_W440},{6,6,0,OFF_W660},{4,0,4,OFF_W404},{4,2,4,OFF_W424},
    {4,2,6,OFF_W426},{6,0,6,OFF_W606},{6,2,4,OFF_W624},{6,2,6,OFF_W626},
    {4,4,4,OFF_W444},{4,4,6,OFF_W446},{4,6,4,OFF_W464},{4,6,6,OFF_W466},
    {6,4,4,OFF_W644},{6,4,6,OFF_W646},{6,6,4,OFF_W664},{6,6,6,OFF_W666}};
  const int l1 = tbl[blockIdx.x][0], l2 = tbl[blockIdx.x][1], l3 = tbl[blockIdx.x][2];
  const int off = tbl[blockIdx.x][3];
  const int d1 = 2 * l1 + 1, d2 = 2 * l2 + 1, d3 = 2 * l3 + 1;
  const int tot = d1 * d2 * d3;
  __shared__ double Cc[2197];
  __shared__ double red[256];
  const int tid = threadIdx.x;

  for (int i = tid; i < tot; i += 256) Cc[i] = 0.0;
  __syncthreads();
  for (int t = tid; t < d1 * d2; t += 256) {
    int i1 = t / d2, i2 = t % d2;
    int m1 = i1 - l1, m2 = i2 - l2, m3 = m1 + m2;
    if (m3 >= -l3 && m3 <= l3)
      Cc[(i1 * d2 + i2) * d3 + (m3 + l3)] = cgcoef(l1, l2, l3, m1, m2, m3);
  }
  __syncthreads();

  auto calcT = [&](int e, double& tre, double& tim) {
    int a = e / (d2 * d3); int rem = e % (d2 * d3); int b = rem / d3; int c = rem % d3;
    int c1[2], c2[2], c3[2]; double r1[2], i1[2], r2[2], i2[2], r3[2], i3[2];
    int n1 = urow(l1, a, c1, r1, i1);
    int n2 = urow(l2, b, c2, r2, i2);
    int n3 = urow(l3, c, c3, r3, i3);
    tre = 0.0; tim = 0.0;
    for (int xx = 0; xx < n1; ++xx)
      for (int yy = 0; yy < n2; ++yy)
        for (int zz = 0; zz < n3; ++zz) {
          double ar = r1[xx], ai = i1[xx], br = r2[yy], bi = i2[yy];
          double cr = r3[zz], ci = -i3[zz];  // conj(U3)
          double pr = ar * br - ai * bi, pi = ar * bi + ai * br;
          double qr = pr * cr - pi * ci, qi = pr * ci + pi * cr;
          double cv = Cc[(c1[xx] * d2 + c2[yy]) * d3 + c3[zz]];
          tre += qr * cv; tim += qi * cv;
        }
  };

  double sre = 0.0, sim = 0.0;
  for (int e = tid; e < tot; e += 256) { double tr, ti; calcT(e, tr, ti); sre += fabs(tr); sim += fabs(ti); }
  red[tid] = sre; __syncthreads();
  for (int s = 128; s > 0; s >>= 1) { if (tid < s) red[tid] += red[tid + s]; __syncthreads(); }
  double totRe = red[0]; __syncthreads();
  red[tid] = sim; __syncthreads();
  for (int s = 128; s > 0; s >>= 1) { if (tid < s) red[tid] += red[tid + s]; __syncthreads(); }
  double totIm = red[0]; __syncthreads();
  const int useRe = (totRe >= totIm) ? 1 : 0;

  double ss = 0.0;
  for (int e = tid; e < tot; e += 256) { double tr, ti; calcT(e, tr, ti); double v = useRe ? tr : ti; ss += v * v; }
  red[tid] = ss; __syncthreads();
  for (int s = 128; s > 0; s >>= 1) { if (tid < s) red[tid] += red[tid + s]; __syncthreads(); }
  double inv = 1.0 / sqrt(red[0]); __syncthreads();

  for (int e = tid; e < tot; e += 256) {
    double tr, ti; calcT(e, tr, ti);
    ws[off + e] = (float)((useRe ? tr : ti) * inv);
  }
}

// ================= setup kernel 2: fold weights + SH =================
__device__ void shvals(int k, double* s2, double& s0) {
  int iy = k / 3, ix = k % 3;
  double kx = (double)(ix - 1), ky = (double)(iy - 1), kz = 0.0;
  if (ix == 1 && iy == 1) { kx = 0.0; ky = 0.0; kz = 1.0; }
  double nn = sqrt(kx * kx + ky * ky + kz * kz);
  if (nn < 1e-8) nn = 1e-8;
  double x = kx / nn, y = ky / nn, z = kz / nn;
  const double s = 0.28209479177387814;  // 1/sqrt(4*pi)
  s0 = s;
  const double r15 = 3.87298334620741688, r5 = 2.23606797749978969;
  s2[0] = r15 * x * z * s;
  s2[1] = r15 * x * y * s;
  s2[2] = r5 * (y * y - 0.5 * (x * x + z * z)) * s;
  s2[3] = r15 * y * z * s;
  s2[4] = 0.5 * r15 * (z * z - x * x) * s;
}

__global__ void __launch_bounds__(256) combine_kernel(float* __restrict__ ws,
                                                      const float* __restrict__ w_score,
                                                      const float* __restrict__ w_val,
                                                      const float* __restrict__ w_out) {
  const int tid = threadIdx.x;
  // S (22x22 block-diagonal score matrix)
  for (int idx = tid; idx < 484; idx += 256) {
    int a = idx / 22, b = idx % 22; double v = 0.0;
    if (a < 9 && b < 9)        v = C_SC * (double)w_score[0] * (double)ws[OFF_W440 + a * 9 + b];
    else if (a >= 9 && b >= 9) v = C_SC * (double)w_score[1] * (double)ws[OFF_W660 + (a - 9) * 13 + (b - 9)];
    ws[OFF_S + idx] = (float)v;
  }
  // Mv[k] : 22x22 value matrices, zero-padded to 32x32
  for (int idx = tid; idx < 9 * 32 * 32; idx += 256) {
    int k = idx / 1024, rem = idx % 1024, a = rem / 32, c = rem % 32;
    double v = 0.0;
    if (a < 22 && c < 22) {
      double s2[5], s0; shvals(k, s2, s0);
      if (a < 9) {
        if (c < 9) {
          double acc = (double)w_val[0] * (double)ws[OFF_W404 + a * 9 + c] * s0;
          for (int b = 0; b < 5; ++b) acc += (double)w_val[1] * (double)ws[OFF_W424 + (a * 5 + b) * 9 + c] * s2[b];
          v = C4V * acc;
        } else {
          int cc = c - 9; double acc = 0.0;
          for (int b = 0; b < 5; ++b) acc += (double)w_val[2] * (double)ws[OFF_W426 + (a * 5 + b) * 13 + cc] * s2[b];
          v = C6V * acc;
        }
      } else {
        int aa = a - 9;
        if (c < 9) {
          double acc = 0.0;
          for (int b = 0; b < 5; ++b) acc += (double)w_val[4] * (double)ws[OFF_W624 + (aa * 5 + b) * 9 + c] * s2[b];
          v = C4V * acc;
        } else {
          int cc = c - 9;
          double acc = (double)w_val[3] * (double)ws[OFF_W606 + aa * 13 + cc] * s0;
          for (int b = 0; b < 5; ++b) acc += (double)w_val[5] * (double)ws[OFF_W626 + (aa * 5 + b) * 13 + cc] * s2[b];
          v = C6V * acc;
        }
      }
    }
    ws[OFF_MV + idx] = (float)v;
  }
  // Gflat : (a*22+b, c) bilinear output tensor, padded to 512x32
  for (int idx = tid; idx < 512 * 32; idx += 256) {
    int r = idx / 32, c = idx % 32; double v = 0.0;
    if (r < 484 && c < 22) {
      int a = r / 22, b = r % 22;
      if (c < 9) {
        double acc;
        if (a < 9) acc = (b < 9) ? (double)w_out[0] * (double)ws[OFF_W444 + (a * 9 + b) * 9 + c]
                                 : (double)w_out[1] * (double)ws[OFF_W464 + (a * 13 + (b - 9)) * 9 + c];
        else       acc = (b < 9) ? (double)w_out[2] * (double)ws[OFF_W644 + ((a - 9) * 9 + b) * 9 + c]
                                 : (double)w_out[3] * (double)ws[OFF_W664 + ((a - 9) * 13 + (b - 9)) * 9 + c];
        v = C4O * acc;
      } else {
        int cc = c - 9; double acc;
        if (a < 9) acc = (b < 9) ? (double)w_out[4] * (double)ws[OFF_W446 + (a * 9 + b) * 13 + cc]
                                 : (double)w_out[5] * (double)ws[OFF_W466 + (a * 13 + (b - 9)) * 13 + cc];
        else       acc = (b < 9) ? (double)w_out[6] * (double)ws[OFF_W646 + ((a - 9) * 9 + b) * 13 + cc]
                                 : (double)w_out[7] * (double)ws[OFF_W666 + ((a - 9) * 13 + (b - 9)) * 13 + cc];
        v = C6O * acc;
      }
    }
    ws[OFF_G + idx] = (float)v;
  }
}

// ================= main kernel =================
__device__ __forceinline__ void wave_lds_sync() {
  // wave-internal LDS write->read ordering: drain DS counter, pin scheduling
  asm volatile("s_wait_dscnt 0" ::: "memory");
  __builtin_amdgcn_wave_barrier();
}

__device__ __forceinline__ v16h frag16(const _Float16* base, int row, int hi) {
  // CDNA5 16-bit A/B fragment: lane<16 -> K{0..7,16..23}; lane>=16 -> K{8..15,24..31}
  v8h a0 = *(const v8h*)(base + row * 32 + hi * 8);
  v8h a1 = *(const v8h*)(base + row * 32 + 16 + hi * 8);
  v16h r;
#pragma unroll
  for (int e = 0; e < 8; ++e) { r[e] = a0[e]; r[8 + e] = a1[e]; }
  return r;
}

__global__ void __launch_bounds__(256) eau_main(const float* __restrict__ f4,
                                                const float* __restrict__ f6,
                                                const float* __restrict__ ws,
                                                float* __restrict__ out) {
  __shared__ __align__(16) _Float16 sMvT[9][32][32];   // [k][n][K] transposed B
  __shared__ __align__(16) _Float16 sGT[22][512];      // [n][K]   transposed B
  __shared__ __align__(16) _Float16 zrow[512];         // zero pad columns
  __shared__ float sS[22][22];
  __shared__ __align__(16) _Float16 sQh[8][16][24];
  __shared__ __align__(16) _Float16 sCtx[8][16][24];
  __shared__ __align__(16) _Float16 sA[8][16][32];     // per-wave A staging

  const int tid = threadIdx.x;
  for (int idx = tid; idx < 9 * 32 * 32; idx += 256) {
    int k = idx / 1024, rem = idx % 1024, a = rem / 32, c = rem % 32;
    sMvT[k][c][a] = (_Float16)ws[OFF_MV + idx];
  }
  for (int idx = tid; idx < 512 * 22; idx += 256) {
    int r = idx / 22, c = idx % 22;
    sGT[c][r] = (_Float16)ws[OFF_G + r * 32 + c];
  }
  for (int idx = tid; idx < 512; idx += 256) zrow[idx] = (_Float16)0.f;
  for (int idx = tid; idx < 484; idx += 256) sS[idx / 22][idx % 22] = ws[OFF_S + idx];
  __syncthreads();

  const int wave = tid >> 5;
  const int lane = tid & 31;
  const int tileBase = blockIdx.x * 128 + wave * 16;
  const int row = lane & 15;
  const int hi = lane >> 4;

  float attn[9];
  int y = 0, x = 0;
  if (lane < 16) {
    const int np = tileBase + lane;
    y = np >> 9; x = np & 511;
    const int src = (y >> 2) * WSRC + (x >> 2);
    float q[22];
#pragma unroll
    for (int c = 0; c < 9; ++c) q[c] = f4[src * 9 + c];
#pragma unroll
    for (int c = 0; c < 13; ++c) q[9 + c] = f6[src * 13 + c];
#pragma unroll
    for (int c = 0; c < 22; ++c) sQh[wave][lane][c] = (_Float16)q[c];
    // T = q * S  (scores left operand)
    float T[22];
#pragma unroll
    for (int b = 0; b < 22; ++b) {
      float acc = 0.f;
#pragma unroll
      for (int a = 0; a < 22; ++a) acc += q[a] * sS[a][b];
      T[b] = acc;
    }
    float mx = -3.0e38f;
#pragma unroll
    for (int k = 0; k < 9; ++k) {
      const int i = k / 3, j = k % 3;
      int yy = y + i - 1; yy = yy < 0 ? 0 : (yy > 511 ? 511 : yy);
      int xx = x + j - 1; xx = xx < 0 ? 0 : (xx > 511 ? 511 : xx);
      const int s2 = (yy >> 2) * WSRC + (xx >> 2);
      float s = 0.f;
#pragma unroll
      for (int c = 0; c < 9; ++c) s += T[c] * f4[s2 * 9 + c];
#pragma unroll
      for (int c = 0; c < 13; ++c) s += T[9 + c] * f6[s2 * 13 + c];
      attn[k] = s; mx = fmaxf(mx, s);
    }
    float sum = 0.f;
#pragma unroll
    for (int k = 0; k < 9; ++k) { attn[k] = __expf(attn[k] - mx); sum += attn[k]; }
    const float inv = 1.0f / sum;
#pragma unroll
    for (int k = 0; k < 9; ++k) attn[k] *= inv;
    // zero K-pad of A staging (columns 22..31) once
#pragma unroll
    for (int c = 22; c < 32; ++c) sA[wave][lane][c] = (_Float16)0.f;
  }

  const _Float16* aBase = &sA[wave][0][0];
  // ---- ctx GEMM: ctx(16x22) = sum_k (attn_k * p_k)(16x22) @ Mv[k](22x22) ----
  v8f c0 = {0.f, 0.f, 0.f, 0.f, 0.f, 0.f, 0.f, 0.f};
  v8f c1 = c0;
#pragma unroll
  for (int k = 0; k < 9; ++k) {
    if (lane < 16) {
      const int i = k / 3, j = k % 3;
      int yy = y + i - 1; yy = yy < 0 ? 0 : (yy > 511 ? 511 : yy);
      int xx = x + j - 1; xx = xx < 0 ? 0 : (xx > 511 ? 511 : xx);
      const int s2 = (yy >> 2) * WSRC + (xx >> 2);
      const float ak = attn[k];
#pragma unroll
      for (int c = 0; c < 9; ++c) sA[wave][lane][c] = (_Float16)(ak * f4[s2 * 9 + c]);
#pragma unroll
      for (int c = 0; c < 13; ++c) sA[wave][lane][9 + c] = (_Float16)(ak * f6[s2 * 13 + c]);
    }
    wave_lds_sync();
    v16h av = frag16(aBase, row, hi);
    const _Float16* mB = &sMvT[k][0][0];
    v16h bv0 = frag16(mB, row, hi);
    v16h bv1 = frag16(mB, 16 + row, hi);
    c0 = __builtin_amdgcn_wmma_f32_16x16x32_f16(false, av, false, bv0, (short)0, c0, false, false);
    c1 = __builtin_amdgcn_wmma_f32_16x16x32_f16(false, av, false, bv1, (short)0, c1, false, false);
    wave_lds_sync();
  }
  // scatter ctx (D layout: lane -> col row, rows r+8*hi)
#pragma unroll
  for (int r = 0; r < 8; ++r) sCtx[wave][r + 8 * hi][row] = (_Float16)c0[r];
  if (row < 6) {
#pragma unroll
    for (int r = 0; r < 8; ++r) sCtx[wave][r + 8 * hi][16 + row] = (_Float16)c1[r];
  }
  wave_lds_sync();

  // ---- output GEMM: o(16x22) = outer(q,ctx)(16x484) @ Gflat(484x22) ----
  v8f o0 = {0.f, 0.f, 0.f, 0.f, 0.f, 0.f, 0.f, 0.f};
  v8f o1 = o0;
  for (int t = 0; t < 16; ++t) {
    // all 32 lanes build 16 outer-product f16 values each
#pragma unroll
    for (int j = 0; j < 16; ++j) {
      const int r = t * 32 + hi * 16 + j;
      _Float16 v = (_Float16)0.f;
      if (r < 484) {
        const int a = r / 22, b = r - a * 22;
        v = (_Float16)((float)sQh[wave][row][a] * (float)sCtx[wave][row][b]);
      }
      sA[wave][row][hi * 16 + j] = v;
    }
    wave_lds_sync();
    v16h av = frag16(aBase, row, hi);
    const _Float16* g0 = &sGT[row][0] + t * 32;
    const _Float16* g1 = (row < 6) ? (&sGT[16 + row][0] + t * 32) : (&zrow[0] + t * 32);
    v16h bv0 = frag16(g0, 0, hi);
    v16h bv1 = frag16(g1, 0, hi);
    o0 = __builtin_amdgcn_wmma_f32_16x16x32_f16(false, av, false, bv0, (short)0, o0, false, false);
    o1 = __builtin_amdgcn_wmma_f32_16x16x32_f16(false, av, false, bv1, (short)0, o1, false, false);
    wave_lds_sync();
  }

  // ---- residual + writeout: d_out = [o4 (N x 9) | o6 (N x 13)] ----
#pragma unroll
  for (int r = 0; r < 8; ++r) {
    const int m = r + 8 * hi;
    const int np = tileBase + m;
    const float v0 = o0[r] + (float)sQh[wave][m][row];
    if (row < 9) out[np * 9 + row] = v0;
    else         out[NTOT * 9 + np * 13 + (row - 9)] = v0;
    if (row < 6) {
      const float v1 = o1[r] + (float)sQh[wave][m][16 + row];
      out[NTOT * 9 + np * 13 + (7 + row)] = v1;
    }
  }
}

extern "C" void kernel_launch(void* const* d_in, const int* in_sizes, int n_in,
                              void* d_out, int out_size, void* d_ws, size_t ws_size,
                              hipStream_t stream) {
  const float* f4 = (const float*)d_in[0];
  const float* f6 = (const float*)d_in[1];
  const float* w_score = (const float*)d_in[2];
  const float* w_val = (const float*)d_in[3];
  const float* w_out = (const float*)d_in[4];
  float* ws = (float*)d_ws;
  float* out = (float*)d_out;

  w3j_kernel<<<16, 256, 0, stream>>>(ws);
  combine_kernel<<<1, 256, 0, stream>>>(ws, w_score, w_val, w_out);
  eau_main<<<NTOT / 128, 256, 0, stream>>>(f4, f6, ws, out);

  (void)in_sizes; (void)n_in; (void)out_size; (void)ws_size;
}
